// SpanEmo_32409823216087
// MI455X (gfx1250) — matrix operand order repaired and checked by reading, not compiled, not run on hardware
//
#include <hip/hip_runtime.h>
#include <hip/hip_bf16.h>
#include <math.h>

typedef __attribute__((ext_vector_type(16))) __bf16 v16bf;
typedef __attribute__((ext_vector_type(8)))  float  v8f;
typedef unsigned int u32x4 __attribute__((ext_vector_type(4)));
typedef int          i32x8 __attribute__((ext_vector_type(8)));
typedef int          i32x4 __attribute__((ext_vector_type(4)));

// Problem constants (from reference): B=256, S=512, H=768, C=11
#define BB   256
#define SS   512
#define HH   768
#define CC   11
#define MM   (BB * CC)          // 2816 rows actually needed
#define MT   (MM / 16)          // 176 M-tiles (exact)
#define KT   (HH / 32)          // 24 K-tiles of 32
#define NT   (HH / 16)          // 48 N-tiles of 16
#define FRAG 512                // halfs per 16x32 / 32x16 bf16 fragment (32 lanes x 16)
#define SLICE_HALFS (NT * FRAG)         // one K-slice of B: 24576 halfs = 48 KB
#define SLICE_DWORDS (SLICE_HALFS / 2)  // 12288 dwords

// ---------------------------------------------------------------------------
// TDM: DMA one contiguous 48KB B K-slice from global into LDS.
// Descriptor bit-packing per CDNA5 ISA §8 (D# groups 0/1); trailing groups
// zero (<=2D tensor). data_size=4B, tile = 12288 x 1 dwords, stride 12288.
// 6-arg builtin form (clang-23 / therock-10.0 headers).
// ---------------------------------------------------------------------------
__device__ __forceinline__ void tdm_load_b_slice(const __bf16* gsrc, unsigned lds_off) {
    unsigned long long ga = (unsigned long long)(uintptr_t)gsrc;
    u32x4 g0;
    g0[0] = 1u;                                            // count=1, user desc
    g0[1] = lds_off;                                       // lds_addr (bytes)
    g0[2] = (unsigned)(ga & 0xFFFFFFFFu);                  // global_addr[31:0]
    g0[3] = (unsigned)((ga >> 32) & 0x01FFFFFFu)           // global_addr[56:32]
          | 0x80000000u;                                   // type=2 ("image")
    i32x8 g1;
    g1[0] = 0x20000;                         // data_size=2 -> 4 bytes; mask=0
    g1[1] = (int)(SLICE_DWORDS << 16);       // tensor_dim0[15:0] in bits 63:48
    g1[2] = 0x10000;                         // tensor_dim1=1 (bits 95:80)
    g1[3] = (int)(SLICE_DWORDS << 16);       // tile_dim0 (bits 127:112)
    g1[4] = 1;                               // tile_dim1=1, tile_dim2=0
    g1[5] = SLICE_DWORDS;                    // tensor_dim0_stride[31:0]
    g1[6] = 0;
    g1[7] = 0;
    i32x4 z4 = {0, 0, 0, 0};
    i32x8 z8 = {0, 0, 0, 0, 0, 0, 0, 0};
    __builtin_amdgcn_tensor_load_to_lds(g0, g1, z4, z4, z8, 0);
}

// ---------------------------------------------------------------------------
// Pack A: gather hidden[b, label_idxs[c], :] rows, convert f32->bf16, WMMA
// A-fragment order (16-bit A 16x32 per ISA 7.12.2).
// ---------------------------------------------------------------------------
__global__ void __launch_bounds__(256)
pack_A(const float* __restrict__ hidden, const int* __restrict__ lidx,
       __bf16* __restrict__ Apack) {
    int tid = blockIdx.x * blockDim.x + threadIdx.x;   // one thread per half
    int j    = tid & 15;
    int lane = (tid >> 4) & 31;
    int fid  = tid >> 9;            // fragment index = mt*KT + kt
    int kt   = fid % KT;
    int mt   = fid / KT;
    int row  = mt * 16 + (lane & 15);
    int bk   = (lane < 16) ? 0 : 8;
    int k    = kt * 32 + ((j < 8) ? (bk + j) : (bk + 16 + (j - 8)));
    int b    = row / CC;
    int c    = row - b * CC;
    int s    = lidx[c];
    float v  = hidden[((long)b * SS + s) * HH + k];
    Apack[tid] = (__bf16)v;
}

// ---------------------------------------------------------------------------
// Pack B: W1 (row-major HxH) -> bf16 B-fragments, K-slice-major layout:
// Bpack[kt][nt][512] so one kt's 48 fragments are contiguous (48KB) for TDM.
// Fragment layout (16-bit B 32x16): lanes 0-15: N=lane, half j -> K=j (j<8), 16+(j-8) (j>=8);
// lanes 16-31: N=lane-16, half j -> K=8+j (j<8), 24+(j-8) (j>=8).
// ---------------------------------------------------------------------------
__global__ void __launch_bounds__(256)
pack_B(const float* __restrict__ W1, __bf16* __restrict__ Bpack) {
    int tid = blockIdx.x * blockDim.x + threadIdx.x;   // total = KT*NT*FRAG
    int j    = tid & 15;
    int lane = (tid >> 4) & 31;
    int fid  = tid >> 9;            // fragment index = kt*NT + nt
    int nt   = fid % NT;
    int kt   = fid / NT;
    int n    = nt * 16 + (lane & 15);
    int k    = kt * 32 + ((j < 8) ? (((lane < 16) ? 0 : 8) + j) : (((lane < 16) ? 0 : 8) + 16 + (j - 8)));
    Bpack[tid] = (__bf16)(W1[(long)k * HH + n]);
}

// ---------------------------------------------------------------------------
// Fused GEMM + tanh + dot(W2). One block per 16-row M-tile, 8 waves x 6
// N-tiles. B K-slices are TDM-DMA'd into a double-buffered 2x48KB LDS region;
// WMMA B operands come from ds_load, A is software-pipelined from global.
// ---------------------------------------------------------------------------
__global__ void __launch_bounds__(256)
gemm_tanh_dot(const __bf16* __restrict__ Apack, const __bf16* __restrict__ Bpack,
              const float* __restrict__ b1, const float* __restrict__ W2,
              const float* __restrict__ b2v, float* __restrict__ logits) {
    __shared__ __attribute__((aligned(64))) __bf16 Bs[2][SLICE_HALFS];  // 96 KB
    __shared__ float s_scores[16];
    const int tid  = threadIdx.x;
    const int wave = tid >> 5;
    const int lane = tid & 31;
    const int mt   = blockIdx.x;

    if (tid < 16) s_scores[tid] = 0.0f;

    const unsigned lds0 = (unsigned)(uintptr_t)&Bs[0][0];
    const unsigned lds1 = (unsigned)(uintptr_t)&Bs[1][0];

    // Prologue: DMA K-slice 0 into buffer 0 (wave 0 owns the TDM queue).
    if (wave == 0) {
        tdm_load_b_slice(Bpack, lds0);
    }

    v8f acc[6] = {};
    const __bf16* aBase = Apack + ((long)mt * KT) * FRAG + lane * 16;
    v16bf a_cur = *(const v16bf*)(aBase);

    for (int kt = 0; kt < KT; ++kt) {
        const int cur = kt & 1;
        if (wave == 0) {
            if (kt + 1 < KT) {
                // Prefetch next K-slice into the other buffer, then wait for
                // the current slice (TENSORcnt is in-order: cnt<=1 => slice
                // kt complete, slice kt+1 may still be in flight).
                tdm_load_b_slice(Bpack + (long)(kt + 1) * SLICE_HALFS,
                                 ((kt + 1) & 1) ? lds1 : lds0);
                __builtin_amdgcn_s_wait_tensorcnt(1);
            } else {
                __builtin_amdgcn_s_wait_tensorcnt(0);
            }
        }
        __syncthreads();   // publish slice kt to all waves

        v16bf a_next = a_cur;
        if (kt + 1 < KT) a_next = *(const v16bf*)(aBase + (long)(kt + 1) * FRAG);

        const __bf16* bs = &Bs[cur][0] + lane * 16;
#pragma unroll
        for (int q = 0; q < 6; ++q) {
            v16bf bf = *(const v16bf*)(bs + (wave * 6 + q) * FRAG);
            acc[q] = __builtin_amdgcn_wmma_f32_16x16x32_bf16(
                false, a_cur, false, bf, (short)0, acc[q], false, false);
        }
        a_cur = a_next;
        __syncthreads();   // all reads of slice kt done before it is re-DMA'd
    }

    // C/D layout: lane holds column n = nt*16 + (lane&15); element i is row
    // (lane<16 ? i : i+8). Apply +b1, tanh, *W2[n], reduce over n into LDS.
    const int col  = lane & 15;
    const int row0 = (lane < 16) ? 0 : 8;
#pragma unroll
    for (int q = 0; q < 6; ++q) {
        const int n = (wave * 6 + q) * 16 + col;
        const float bias = b1[n];
        const float w2   = W2[n];
#pragma unroll
        for (int i = 0; i < 8; ++i) {
            float h = tanhf(acc[q][i] + bias);
            atomicAdd(&s_scores[row0 + i], h * w2);   // ds_add_f32
        }
    }
    __syncthreads();

    if (tid < 16) {
        logits[mt * 16 + tid] = s_scores[tid] + b2v[0];
    }
}

// ---------------------------------------------------------------------------
// Loss: thread b handles row b of the (256,11) logits. Stable BCE, sigmoid,
// corr-loss terms, block tree-reduction, final scalar + y_pred outputs.
// ---------------------------------------------------------------------------
__global__ void __launch_bounds__(256)
loss_kernel(const float* __restrict__ logits, const int* __restrict__ targets,
            float* __restrict__ out) {
    __shared__ float s_bce[256];
    __shared__ float s_corr[256];
    const int b = threadIdx.x;

    float bce = 0.0f;
    float sz = 0.0f, so = 0.0f, nmz = 0.0f, nmo = 0.0f;
#pragma unroll
    for (int c = 0; c < CC; ++c) {
        float x = logits[b * CC + c];
        float t = (float)targets[b * CC + c];
        bce += fmaxf(x, 0.0f) + log1pf(expf(-fabsf(x))) - x * t;  // logaddexp(0,x)-x*t
        float sig = 1.0f / (1.0f + expf(-x));
        sz  += expf(sig) * (1.0f - t);
        so  += expf(-sig) * t;
        nmz += 1.0f - t;
        nmo += t;
        out[1 + b * CC + c] = (x > 0.0f) ? 1.0f : 0.0f;   // sigmoid(x)>0.5
    }
    float corr = sz * so / (nmz * nmo + 1e-7f);

    s_bce[b]  = bce;
    s_corr[b] = corr;
    __syncthreads();
#pragma unroll
    for (int off = 128; off > 0; off >>= 1) {
        if (b < off) {
            s_bce[b]  += s_bce[b + off];
            s_corr[b] += s_corr[b + off];
        }
        __syncthreads();
    }
    if (b == 0) {
        float bce_mean  = s_bce[0]  / (float)(BB * CC);
        float corr_mean = s_corr[0] / (float)BB;
        out[0] = 0.8f * bce_mean + 0.2f * corr_mean;   // (1-ALPHA)*bce + ALPHA*cl
    }
}

// ---------------------------------------------------------------------------
extern "C" void kernel_launch(void* const* d_in, const int* in_sizes, int n_in,
                              void* d_out, int out_size, void* d_ws, size_t ws_size,
                              hipStream_t stream) {
    const float* hidden  = (const float*)d_in[0];
    const float* W1      = (const float*)d_in[1];
    const float* b1      = (const float*)d_in[2];
    const float* W2      = (const float*)d_in[3];
    const float* b2      = (const float*)d_in[4];
    const int*   targets = (const int*)d_in[5];
    const int*   lidx    = (const int*)d_in[6];
    (void)in_sizes; (void)n_in; (void)out_size; (void)ws_size;

    // Workspace layout:
    //   Apack: MT*KT*FRAG bf16 = 4,325,376 B
    //   Bpack: KT*NT*FRAG bf16 = 1,179,648 B  (K-slice-major for TDM)
    //   logits: MM f32         =    11,264 B
    char* ws = (char*)d_ws;
    __bf16* Apack  = (__bf16*)ws;
    __bf16* Bpack  = (__bf16*)(ws + (size_t)MT * KT * FRAG * 2);
    float*  logits = (float*)(ws + (size_t)MT * KT * FRAG * 2
                                 + (size_t)KT * NT * FRAG * 2);

    const int packA_threads = MT * KT * FRAG;   // 2,162,688 -> 8448 blocks
    const int packB_threads = KT * NT * FRAG;   //   589,824 -> 2304 blocks

    pack_A<<<packA_threads / 256, 256, 0, stream>>>(hidden, lidx, Apack);
    pack_B<<<packB_threads / 256, 256, 0, stream>>>(W1, Bpack);
    gemm_tanh_dot<<<MT, 256, 0, stream>>>(Apack, Bpack, b1, W2, b2, logits);
    loss_kernel<<<1, 256, 0, stream>>>(logits, targets, (float*)d_out);
}
